// VariationalCodebook_68874095559159
// MI455X (gfx1250) — compile-verified, matrix operand order checked by reading, not run on hardware
//
#include <hip/hip_runtime.h>

// ---------------- problem constants (from reference) ----------------
constexpr int BB   = 16;
constexpr int TT   = 2048;
constexpr int DD   = 512;     // feature dim (GEMM reduction dim)
constexpr int KK   = 8192;    // codebook entries
constexpr int NQ   = BB * TT; // 32768 query rows
constexpr int TOPK = 8;

// ---------------- CDNA5 async global->LDS availability ----------------
#define USE_ASYNC_LDS 0
#if defined(__has_builtin)
#if __has_builtin(__builtin_amdgcn_global_load_async_to_lds_b128) && \
    __has_builtin(__builtin_amdgcn_s_wait_asynccnt)
#undef USE_ASYNC_LDS
#define USE_ASYNC_LDS 1
#endif
#endif

// ---------------- vector types ----------------
typedef __bf16 bf16x8  __attribute__((ext_vector_type(8)));
typedef __bf16 bf16x16 __attribute__((ext_vector_type(16)));
typedef float  f32x4   __attribute__((ext_vector_type(4)));
typedef float  f32x8   __attribute__((ext_vector_type(8)));
typedef int    v4i     __attribute__((vector_size(16)));   // builtin's b128 unit

struct BF16Pair { bf16x8 lo; bf16x8 hi; };

static __device__ __forceinline__ bf16x8 cvt8(const float* p) {
  bf16x8 r;
#pragma unroll
  for (int i = 0; i < 8; ++i) r[i] = (__bf16)p[i];
  return r;
}

// two contiguous 16B loads -> one WMMA 16-bit operand (per-lane half-rows)
static __device__ __forceinline__ bf16x16 load_b16(const __bf16* p) {
  BF16Pair s;
  s.lo = *reinterpret_cast<const bf16x8*>(p);
  s.hi = *reinterpret_cast<const bf16x8*>(p + 16);
  return __builtin_bit_cast(bf16x16, s);
}

static __device__ __forceinline__ void topk_insert(float (&bv)[TOPK], int (&bi)[TOPK],
                                                   float t, int idx) {
  if (t > bv[TOPK - 1]) {
    bv[TOPK - 1] = t; bi[TOPK - 1] = idx;
#pragma unroll
    for (int j = TOPK - 1; j > 0; --j) {
      if (bv[j] > bv[j - 1]) {
        float tv = bv[j]; bv[j] = bv[j - 1]; bv[j - 1] = tv;
        int   ti = bi[j]; bi[j] = bi[j - 1]; bi[j - 1] = ti;
      }
    }
  }
}

#if USE_ASYNC_LDS
// 16-byte async DMA: global -> LDS, tracked by ASYNCcnt (no VGPR return)
static __device__ __forceinline__ void async_cp16(const __bf16* g, __bf16* l) {
  __builtin_amdgcn_global_load_async_to_lds_b128(
      (__attribute__((address_space(1))) v4i*)(g),
      (__attribute__((address_space(3))) v4i*)(l), 0, 0);
}
constexpr int BPAD  = 528;                 // 512 + 16 pad -> 264-dword row stride
constexpr int ARENA = 2 * 16 * BPAD * 2;   // double-buffered 16-row bf16 tile (33792 B)
#else
constexpr int ARENA = 16384;               // merge scratch only
#endif

// ---------------- kernel 1: codebook -> bf16 + ||c||^2 ----------------
__global__ __launch_bounds__(256)
void vcq_prep(const float* __restrict__ codebook, const int* __restrict__ modality,
              unsigned short* __restrict__ cb_bf16_raw, float* __restrict__ cnorm) {
  const int lane = threadIdx.x & 31;
  const int wave = threadIdx.x >> 5;
  const int k    = blockIdx.x * 8 + wave;   // 0..8191
  const float* src = codebook + ((size_t)modality[0] * KK + k) * DD + lane * 16;
  __bf16*      dst = (__bf16*)cb_bf16_raw + (size_t)k * DD + lane * 16;

  float ss = 0.f;
  bf16x8 o0, o1;
#pragma unroll
  for (int i = 0; i < 8; ++i) { float x = src[i];     ss += x * x; o0[i] = (__bf16)x; }
#pragma unroll
  for (int i = 0; i < 8; ++i) { float x = src[8 + i]; ss += x * x; o1[i] = (__bf16)x; }
  *reinterpret_cast<bf16x8*>(dst)     = o0;
  *reinterpret_cast<bf16x8*>(dst + 8) = o1;

#pragma unroll
  for (int off = 16; off > 0; off >>= 1) ss += __shfl_xor(ss, off, 32);
  if (lane == 0) cnorm[k] = ss;
}

// ---------------- kernel 2: zero the [N,K] weights output (streaming NT) ----------------
__global__ __launch_bounds__(256)
void vcq_zero(f32x4* __restrict__ p, size_t n4) {
  size_t i      = (size_t)blockIdx.x * blockDim.x + threadIdx.x;
  size_t stride = (size_t)gridDim.x * blockDim.x;
  f32x4 z = {};
  for (; i < n4; i += stride) __builtin_nontemporal_store(z, &p[i]);
}

// ---------------- kernel 3: fused dist-GEMM + top-8 + softmax + gather ----------------
__global__ __launch_bounds__(256)
void vcq_main(const float* __restrict__ query,
              const float* __restrict__ completeness,
              const float* __restrict__ codebook,
              const float* __restrict__ log_temp,
              const int*   __restrict__ modality,
              const unsigned short* __restrict__ cb_bf16_raw,
              const float* __restrict__ cnorm,
              float* __restrict__ out_ret,
              float* __restrict__ out_w) {
  constexpr int WAVES = 8;
  __shared__ float tileS[WAVES][16][17];                 // score-tile transpose staging
  __shared__ __align__(16) unsigned char arena[ARENA];   // B staging / merge scratch

  const int lane  = threadIdx.x & 31;
  const int wave  = threadIdx.x >> 5;
  const int m     = lane & 15;                // A row / B col / C col owned by this lane
  const int g     = lane >> 4;                // half-wave selector (K striping)
  const int qtile = blockIdx.x * WAVES + wave;
  const int row0  = qtile * 16;

  const __bf16* cbbf = (const __bf16*)cb_bf16_raw;

#if USE_ASYNC_LDS
  __bf16* ldsB = reinterpret_cast<__bf16*>(arena);       // [2][16][BPAD]
  // block cooperatively DMAs tile 0 (16 codes x 512 bf16 = 16KB): 64B per thread
  {
    const int trow = threadIdx.x >> 4;            // 0..15 (code row in tile)
    const int tcol = (threadIdx.x & 15) * 32;     // element offset of 64B chunk
    const __bf16* src = cbbf + (size_t)trow * DD + tcol;
    __bf16*       dst = ldsB + (size_t)trow * BPAD + tcol;
    async_cp16(src,      dst);      async_cp16(src +  8, dst +  8);
    async_cp16(src + 16, dst + 16); async_cp16(src + 24, dst + 24);
  }
#endif

  // ---- load + convert the 16x512 query tile into WMMA A layout (overlaps DMA) ----
  bf16x16 aReg[16];
  {
    const float* qrow = query + (size_t)(row0 + m) * DD;
#pragma unroll
    for (int kk = 0; kk < 16; ++kk) {
      const float* p0 = qrow + kk * 32 + g * 8;
      BF16Pair s { cvt8(p0), cvt8(p0 + 16) };
      aReg[kk] = __builtin_bit_cast(bf16x16, s);
    }
  }

#if USE_ASYNC_LDS
  __builtin_amdgcn_s_wait_asynccnt(0);
  __syncthreads();
#endif

  // ---- running top-8 of t = 2*dot - ||c||^2 (rank-equivalent to sim) ----
  float bv[TOPK]; int bi[TOPK];
#pragma unroll
  for (int j = 0; j < TOPK; ++j) { bv[j] = -__builtin_huge_valf(); bi[j] = 0; }

  for (int c0 = 0; c0 < KK; c0 += 16) {
    if (c0 + 32 < KK)  // keep tile-after-next flowing toward L1 (global_prefetch_b8)
      __builtin_prefetch((const void*)(cbbf + (size_t)(c0 + 32 + m) * DD + g * 8), 0, 1);

#if USE_ASYNC_LDS
    const int cur = (c0 >> 4) & 1;
    if (c0 + 16 < KK) {   // kick DMA for the next tile into the other buffer
      const int  trow = threadIdx.x >> 4;
      const int  tcol = (threadIdx.x & 15) * 32;
      const __bf16* src = cbbf + (size_t)(c0 + 16 + trow) * DD + tcol;
      __bf16*       dst = ldsB + (size_t)((cur ^ 1) * 16 + trow) * BPAD + tcol;
      async_cp16(src,      dst);      async_cp16(src +  8, dst +  8);
      async_cp16(src + 16, dst + 16); async_cp16(src + 24, dst + 24);
    }
    const __bf16* bptr = ldsB + (size_t)(cur * 16 + m) * BPAD + g * 8;
#else
    const __bf16* bptr = cbbf + (size_t)(c0 + m) * DD + g * 8;
#endif

    f32x8 acc = {};
#pragma unroll
    for (int kk = 0; kk < 16; ++kk) {
      bf16x16 b = load_b16(bptr + kk * 32);
      acc = __builtin_amdgcn_wmma_f32_16x16x32_bf16(
                false, aReg[kk], false, b, (short)0, acc, false, false);
    }

    // fold -||c||^2 and transpose through LDS: write (row = j+8g, col = m)
    const float cn = cnorm[c0 + m];
#pragma unroll
    for (int j = 0; j < 8; ++j)
      tileS[wave][j + 8 * g][m] = 2.0f * acc[j] - cn;
    __syncthreads();

    // lane owns query row m; scans its half of the 16 candidate columns
#pragma unroll
    for (int jj = 0; jj < 8; ++jj) {
      float t = tileS[wave][m][g * 8 + jj];
      topk_insert(bv, bi, t, c0 + g * 8 + jj);
    }
#if USE_ASYNC_LDS
    __builtin_amdgcn_s_wait_asynccnt(0);   // next buffer fully landed
#endif
    __syncthreads();
  }

  // ---- merge half-wave lists via arena scratch (B staging is dead now) ----
  float* pvS = reinterpret_cast<float*>(arena);           // [8][16][8]
  int*   piS = reinterpret_cast<int*>(pvS + WAVES * 16 * TOPK);
  float* wvS = reinterpret_cast<float*>(piS + WAVES * 16 * TOPK);
  int*   wiS = reinterpret_cast<int*>(wvS + WAVES * 16 * TOPK);
  const int rowSlot = (wave * 16 + m) * TOPK;

  if (g == 1) {
#pragma unroll
    for (int j = 0; j < TOPK; ++j) { pvS[rowSlot + j] = bv[j]; piS[rowSlot + j] = bi[j]; }
  }
  __syncthreads();

  if (g == 0) {
#pragma unroll
    for (int j = 0; j < TOPK; ++j) topk_insert(bv, bi, pvS[rowSlot + j], piS[rowSlot + j]);

    // softmax with adaptive temperature (q-norm cancels under max-subtraction)
    const int   r     = row0 + m;
    const float comp  = completeness[(unsigned)r / TT];
    const float temp  = __expf(log_temp[0]);
    const float scale = 1.0f / (sqrtf((float)DD) * (temp * (2.0f - comp)));
    const float mx    = bv[0];                 // sorted descending
    float e[TOPK], s = 0.f;
#pragma unroll
    for (int j = 0; j < TOPK; ++j) { e[j] = __expf((bv[j] - mx) * scale); s += e[j]; }
    const float inv = 1.0f / s;

    float* wrow = out_w + (size_t)r * KK;
#pragma unroll
    for (int j = 0; j < TOPK; ++j) {
      const float w = e[j] * inv;
      wvS[rowSlot + j] = w; wiS[rowSlot + j] = bi[j];
      wrow[bi[j]] = w;                         // sparse scatter into zeroed region
    }
  }
  __syncthreads();

  // ---- cooperative retrieved = sum_j w_j * cb_f32[idx_j], coalesced per row ----
  const float* cbf = codebook + (size_t)modality[0] * KK * DD;
  for (int mm = 0; mm < 16; ++mm) {
    const int r    = row0 + mm;
    const int slot = (wave * 16 + mm) * TOPK;
    f32x4 a0 = {}, a1 = {}, a2 = {}, a3 = {};
#pragma unroll
    for (int j = 0; j < TOPK; ++j) {
      const float  w  = wvS[slot + j];
      const f32x4* c4 = reinterpret_cast<const f32x4*>(cbf + (size_t)wiS[slot + j] * DD)
                        + lane * 4;           // wave covers the whole 2KB row contiguously
      a0 += w * c4[0]; a1 += w * c4[1]; a2 += w * c4[2]; a3 += w * c4[3];
    }
    f32x4* o4 = reinterpret_cast<f32x4*>(out_ret + (size_t)r * DD) + lane * 4;
    __builtin_nontemporal_store(a0, o4 + 0);
    __builtin_nontemporal_store(a1, o4 + 1);
    __builtin_nontemporal_store(a2, o4 + 2);
    __builtin_nontemporal_store(a3, o4 + 3);
  }
}

// ---------------- host-side launch ----------------
extern "C" void kernel_launch(void* const* d_in, const int* in_sizes, int n_in,
                              void* d_out, int out_size, void* d_ws, size_t ws_size,
                              hipStream_t stream) {
  const float* query        = (const float*)d_in[0];
  const float* completeness = (const float*)d_in[1];
  const float* codebook     = (const float*)d_in[2];
  const float* log_temp     = (const float*)d_in[3];
  const int*   modality     = (const int*)d_in[4];
  // d_in[5] = top_k (compile-time TOPK=8)

  float* out_ret = (float*)d_out;
  float* out_w   = out_ret + (size_t)NQ * DD;

  unsigned short* cb_bf16 = (unsigned short*)d_ws;                       // 8 MB
  float*          cnorm   = (float*)((char*)d_ws + (size_t)KK * DD * 2); // +32 KB

  vcq_prep<<<KK / 8, 256, 0, stream>>>(codebook, modality, cb_bf16, cnorm);
  vcq_zero<<<2048, 256, 0, stream>>>((f32x4*)out_w, (size_t)NQ * KK / 4);
  vcq_main<<<NQ / 128, 256, 0, stream>>>(query, completeness, codebook, log_temp,
                                         modality, cb_bf16, cnorm, out_ret, out_w);
}
